// LuongConcatAttention_67568425501583
// MI455X (gfx1250) — matrix operand order repaired and checked by reading, not compiled, not run on hardware
//
#include <hip/hip_runtime.h>
#include <hip/hip_bf16.h>
#include <math.h>

#define B_SEG 64
#define N_TOK 65536
#define HDIM  512

typedef __attribute__((ext_vector_type(16))) __bf16 v16bf;
typedef __attribute__((ext_vector_type(8)))  __bf16 v8bf;
typedef __attribute__((ext_vector_type(4)))  __bf16 v4bf;
typedef __attribute__((ext_vector_type(8)))  float  v8f;

#define PITCH 520                 // bf16 elems per staged row (1040 B, 16B-aligned, bank-staggered)
#define WAVES_PER_BLOCK 2
#define ROWS_PER_BLOCK (WAVES_PER_BLOCK * 16)

static __device__ __forceinline__ __bf16 f2bf(float f) { return (__bf16)f; }

// ---------------- P = prev @ W1^T + b   (B x H, tiny) ----------------
__global__ void prep_P_kernel(const float* __restrict__ prev,
                              const float* __restrict__ W,
                              const float* __restrict__ bias,
                              float* __restrict__ P)
{
    int id = blockIdx.x * blockDim.x + threadIdx.x;   // 0 .. B*H-1
    int bb = id >> 9;
    int h  = id & (HDIM - 1);
    const float* pr = prev + (size_t)bb * HDIM;
    const float* wr = W + (size_t)h * (2 * HDIM);     // W1 row h = W[h, 0:512]
    float s = 0.f;
    #pragma unroll 4
    for (int k = 0; k < HDIM; k += 4) {
        float4 a = *(const float4*)(pr + k);
        float4 w = *(const float4*)(wr + k);
        s += a.x * w.x + a.y * w.y + a.z * w.z + a.w * w.w;
    }
    P[id] = s + bias[h];
}

// ---------------- Wb[h][k] = bf16(W2[h][k]) = bf16(W[h, 512+k]) ----------------
__global__ void prep_Wb_kernel(const float* __restrict__ W, __bf16* __restrict__ Wb)
{
    int id  = blockIdx.x * blockDim.x + threadIdx.x;  // H*H/4 threads
    int idx = id * 4;
    int h = idx >> 9;
    int k = idx & (HDIM - 1);
    float4 f = *(const float4*)(W + (size_t)h * (2 * HDIM) + HDIM + k);
    v4bf o;
    o[0] = f2bf(f.x); o[1] = f2bf(f.y); o[2] = f2bf(f.z); o[3] = f2bf(f.w);
    *(v4bf*)(Wb + idx) = o;
}

// ---------------- main fused kernel: scores = tanh(E@W2^T + P[seg]) @ v ----------------
__global__ void __launch_bounds__(WAVES_PER_BLOCK * 32)
attn_scores_kernel(const float*  __restrict__ enc,     // N x H f32
                   const int*    __restrict__ seg,     // N
                   const __bf16* __restrict__ Wb,      // H x H bf16 (row h = W2[h][:])
                   const float*  __restrict__ P,       // B x H (prev@W1^T + b)
                   const float*  __restrict__ vvec,    // H
                   float*        __restrict__ scores)  // N
{
    extern __shared__ __bf16 lds[];
    const int lane = threadIdx.x & 31;
    const int wave = threadIdx.x >> 5;
    const int r0   = blockIdx.x * ROWS_PER_BLOCK + wave * 16;
    __bf16* myA = lds + wave * 16 * PITCH;

    // Stage this wave's 16 rows x 512 cols of E into LDS as bf16 (coalesced float4 loads).
    #pragma unroll 4
    for (int t = 0; t < 64; ++t) {
        int g   = t * 32 + lane;          // float4 index within 16x128-float4 tile
        int row = g >> 7;
        int c4  = g & 127;
        const float4 f = *(const float4*)(enc + (size_t)(r0 + row) * HDIM + c4 * 4);
        v4bf o;
        o[0] = f2bf(f.x); o[1] = f2bf(f.y); o[2] = f2bf(f.z); o[3] = f2bf(f.w);
        *(v4bf*)(myA + row * PITCH + c4 * 4) = o;
    }
    __syncthreads();

    const int hi   = lane & 15;           // A row / B+C column within tile
    const int half = (lane >> 4) & 1;     // lane group
    const int mb   = half * 8;            // C rows owned by this lane: mb..mb+7

    int sgid[8];
    #pragma unroll
    for (int i = 0; i < 8; ++i) sgid[i] = seg[r0 + mb + i];

    // 16-bit A lane layout: lane<16 holds row=lane, K {0..7,16..23}; lane>=16: K {8..15,24..31}
    const __bf16* ap = myA + hi * PITCH + half * 8;
    float sc[8] = {0.f, 0.f, 0.f, 0.f, 0.f, 0.f, 0.f, 0.f};

    for (int ht = 0; ht < HDIM / 16; ++ht) {
        const int h = ht * 16 + hi;
        // 16-bit B (32x16) lane layout: column = lane&15, K = (lane<16 ? {0..7,16..23} : {8..15,24..31})
        const __bf16* bp = Wb + (size_t)h * HDIM + half * 8;
        v8f acc = {0.f, 0.f, 0.f, 0.f, 0.f, 0.f, 0.f, 0.f};
        #pragma unroll
        for (int kt = 0; kt < HDIM / 32; ++kt) {
            v8bf a0 = *(const v8bf*)(ap + kt * 32);
            v8bf a1 = *(const v8bf*)(ap + kt * 32 + 16);
            v8bf b0 = *(const v8bf*)(bp + kt * 32);
            v8bf b1 = *(const v8bf*)(bp + kt * 32 + 16);
            v16bf a = __builtin_shufflevector(a0, a1, 0,1,2,3,4,5,6,7,8,9,10,11,12,13,14,15);
            v16bf b = __builtin_shufflevector(b0, b1, 0,1,2,3,4,5,6,7,8,9,10,11,12,13,14,15);
            acc = __builtin_amdgcn_wmma_f32_16x16x32_bf16(
                      false, a, false, b, (short)0, acc, false, false);
        }
        // Fused epilogue: energy = tanh(acc + P[seg][h]); score partial += energy * v[h]
        const float vh = vvec[h];
        #pragma unroll
        for (int i = 0; i < 8; ++i) {
            float e = tanhf(acc[i] + P[sgid[i] * HDIM + h]);
            sc[i] += e * vh;
        }
    }

    // Reduce partial scores across the 16 lanes that share the same C rows.
    #pragma unroll
    for (int off = 1; off < 16; off <<= 1) {
        #pragma unroll
        for (int i = 0; i < 8; ++i) sc[i] += __shfl_xor(sc[i], off, 32);
    }
    if (hi == 0) {
        #pragma unroll
        for (int i = 0; i < 8; ++i) scores[r0 + mb + i] = sc[i];
    }
}

// ---------------- segment softmax (tiny passes over 256 KB) ----------------
__global__ void seg_init_kernel(unsigned* __restrict__ segmax, float* __restrict__ denom)
{
    int t = threadIdx.x;
    if (t < B_SEG) { segmax[t] = 0u; denom[t] = 0.f; }
}

__global__ void seg_max_kernel(const float* __restrict__ scores,
                               const int* __restrict__ seg,
                               unsigned* __restrict__ segmax)
{
    int n = blockIdx.x * blockDim.x + threadIdx.x;
    unsigned u = __float_as_uint(scores[n]);
    unsigned o = (u & 0x80000000u) ? ~u : (u | 0x80000000u);   // order-preserving map
    atomicMax(&segmax[seg[n]], o);
}

__global__ void seg_exp_kernel(const float* __restrict__ scores,
                               const int* __restrict__ seg,
                               const unsigned* __restrict__ segmax,
                               float* __restrict__ ebuf,
                               float* __restrict__ denom)
{
    int n = blockIdx.x * blockDim.x + threadIdx.x;
    unsigned o  = segmax[seg[n]];
    unsigned ub = (o & 0x80000000u) ? (o & 0x7FFFFFFFu) : ~o;
    float m  = __uint_as_float(ub);
    float ev = expf(scores[n] - m);
    ebuf[n] = ev;
    atomicAdd(&denom[seg[n]], ev);
}

__global__ void seg_norm_kernel(const float* __restrict__ ebuf,
                                const int* __restrict__ seg,
                                const float* __restrict__ denom,
                                float* __restrict__ out)
{
    int n = blockIdx.x * blockDim.x + threadIdx.x;
    out[n] = ebuf[n] / denom[seg[n]];
}

extern "C" void kernel_launch(void* const* d_in, const int* in_sizes, int n_in,
                              void* d_out, int out_size, void* d_ws, size_t ws_size,
                              hipStream_t stream)
{
    const float* prev = (const float*)d_in[0];   // (B,H)
    const float* enc  = (const float*)d_in[1];   // (N,H)
    const int*   seg  = (const int*)  d_in[2];   // (N,)
    const float* W    = (const float*)d_in[3];   // (H,2H)
    const float* bias = (const float*)d_in[4];   // (H,)
    const float* vv   = (const float*)d_in[5];   // (H,1)
    float* out = (float*)d_out;                  // (N,1)

    char* w = (char*)d_ws;
    float*    P      = (float*)w;    w += (size_t)B_SEG * HDIM * sizeof(float);
    __bf16*   Wb     = (__bf16*)w;   w += (size_t)HDIM * HDIM * sizeof(__bf16);
    float*    scores = (float*)w;    w += (size_t)N_TOK * sizeof(float);
    float*    ebuf   = (float*)w;    w += (size_t)N_TOK * sizeof(float);
    unsigned* segmax = (unsigned*)w; w += (size_t)B_SEG * sizeof(unsigned);
    float*    denom  = (float*)w;    w += (size_t)B_SEG * sizeof(float);

    prep_P_kernel <<<(B_SEG * HDIM) / 256, 256, 0, stream>>>(prev, W, bias, P);
    prep_Wb_kernel<<<(HDIM * HDIM / 4) / 256, 256, 0, stream>>>(W, Wb);
    seg_init_kernel<<<1, 64, 0, stream>>>(segmax, denom);

    size_t ldsbytes = (size_t)WAVES_PER_BLOCK * 16 * PITCH * sizeof(__bf16);  // 33280 B
    attn_scores_kernel<<<N_TOK / ROWS_PER_BLOCK, WAVES_PER_BLOCK * 32, ldsbytes, stream>>>(
        enc, seg, Wb, P, vv, scores);

    seg_max_kernel <<<N_TOK / 256, 256, 0, stream>>>(scores, seg, segmax);
    seg_exp_kernel <<<N_TOK / 256, 256, 0, stream>>>(scores, seg, segmax, ebuf, denom);
    seg_norm_kernel<<<N_TOK / 256, 256, 0, stream>>>(ebuf, seg, denom, out);
}
